// ExampleBitmapGFN_71090298684078
// MI455X (gfx1250) — compile-verified, matrix operand order checked by reading, not run on hardware
//
#include <hip/hip_runtime.h>
#include <math.h>
#include <stdint.h>

#define NEG (-100000000.0f)

typedef __attribute__((ext_vector_type(16))) __bf16 v16bf;
typedef __attribute__((ext_vector_type(8)))  __bf16 v8bf;
typedef __attribute__((ext_vector_type(8)))  float  v8f;

union V16U { v16bf v; v8bf h[2]; };

// Async global -> LDS 16-byte copy (CDNA5 GLOBAL_LOAD_ASYNC_TO_LDS_B128,
// tracked by ASYNCcnt). VDST = VGPR holding LDS byte address, VADDR = 64-bit
// global address. Inline asm: portable across ROCm7.2 / amdgpu-toolchain
// (the clang builtin arity differs between them).
__device__ __forceinline__ void async_copy_b128(void* lds, const void* gptr) {
  uint32_t laddr = (uint32_t)(uintptr_t)lds;   // low 32 bits = LDS offset
  asm volatile("global_load_async_to_lds_b128 %0, %1, off"
               :: "v"(laddr), "v"(gptr) : "memory");
}
__device__ __forceinline__ void wait_asynccnt0() {
  asm volatile("s_wait_asynccnt 0x0" ::: "memory");
}

// ---------------------------------------------------------------------------
// Block-tiled WMMA GEMM with async-LDS staging of B:
//   C[M,N] = op(A)[M,K] @ BT[N,K]^T + bias
// Block = 256 threads (8 waves). Block tile = 128 rows x 64 cols.
// Wave w owns rows [mb*128 + w*16, +16) and all 4 column tiles.
// Per 32-wide k-chunk, the 64x32 bf16 B strip (4KB) is staged into LDS by one
// async b128 per thread, double buffered; all 8 waves consume it via ds loads
// in the ISA 16-bit operand striping (lane<16: K{0..7,16..23}, lane>=16:
// K{8..15,24..31} of its row).
//   A_IS_F32_RELU: A is f32; relu + cvt to bf16 fused into the operand load.
//   OUT_BF16_RELU: bias+relu, store bf16; else store f32 (+bias).
// ---------------------------------------------------------------------------
template<bool A_IS_F32_RELU, bool OUT_BF16_RELU>
__global__ __launch_bounds__(256) void gemm_wmma_lds(
    const void* __restrict__ Av, const __bf16* __restrict__ BT,
    const float* __restrict__ bias, int nbias,
    void* __restrict__ Cv, int M, int N, int K)
{
  __shared__ alignas(16) __bf16 sB[2][64 * 32];

  const int tid  = threadIdx.x;
  const int wid  = tid >> 5;           // wave in block: 0..7
  const int lane = tid & 31;
  const int mblocks = M >> 7;          // 128 rows per block
  const int mb = blockIdx.x % mblocks;
  const int ns = blockIdx.x / mblocks;
  const int col0 = ns * 64;
  if (col0 >= N) return;

  const int half = lane >> 4;
  const int lrow = lane & 15;
  const int arow = mb * 128 + wid * 16 + lrow;

  const float*  Af = (const float*)Av;
  const __bf16* Ab = (const __bf16*)Av;

  // B staging: thread -> one b128 (8 bf16) of the 64x32 strip
  const int sr = tid >> 2;             // strip row 0..63
  const int sk = (tid & 3) << 3;       // k sub-offset 0/8/16/24
  const __bf16* gB = BT + (size_t)(col0 + sr) * K + sk;
  __bf16* lB0 = &sB[0][sr * 32 + sk];
  __bf16* lB1 = &sB[1][sr * 32 + sk];

  v8f acc[4];
#pragma unroll
  for (int t = 0; t < 4; ++t)
#pragma unroll
    for (int e = 0; e < 8; ++e) acc[t][e] = 0.0f;

  // prologue: stage chunk 0
  async_copy_b128(lB0, gB);

  int cur = 0;
  for (int kk = 0; kk < K; kk += 32) {
    wait_asynccnt0();      // own async stores to LDS complete
    __syncthreads();       // ...across all waves; prev reads also complete
    if (kk + 32 < K)
      async_copy_b128(cur ? lB0 : lB1, gB + (kk + 32));

    // ---- A operand (16x32 tile of this wave's rows) ----
    const int k0 = kk + 8 * half;
    V16U a;
    if (A_IS_F32_RELU) {
      const float* ap = Af + (size_t)arow * K + k0;
      __builtin_prefetch(ap + 32, 0, 0);
      v8f x0 = *(const v8f*)(ap);
      v8f x1 = *(const v8f*)(ap + 16);
#pragma unroll
      for (int e = 0; e < 8; ++e) a.v[e]     = (__bf16)fmaxf(x0[e], 0.0f);
#pragma unroll
      for (int e = 0; e < 8; ++e) a.v[8 + e] = (__bf16)fmaxf(x1[e], 0.0f);
    } else {
      const __bf16* ap = Ab + (size_t)arow * K + k0;
      __builtin_prefetch(ap + 32, 0, 0);
      a.h[0] = *(const v8bf*)(ap);
      a.h[1] = *(const v8bf*)(ap + 16);
    }

    // ---- 4 column tiles: B from LDS, then WMMA ----
    const __bf16* bb = &sB[cur][0];
#pragma unroll
    for (int t = 0; t < 4; ++t) {
      const __bf16* bp = bb + (t * 16 + lrow) * 32 + 8 * half;
      V16U b;
      b.h[0] = *(const v8bf*)(bp);
      b.h[1] = *(const v8bf*)(bp + 16);
      acc[t] = __builtin_amdgcn_wmma_f32_16x16x32_bf16(
          false, a.v, false, b.v, (short)0, acc[t], false, false);
    }
    cur ^= 1;
  }

  // C/D layout: n = lane&15 ; m = vgpr + 8*(lane>>4)
#pragma unroll
  for (int t = 0; t < 4; ++t) {
    const int col = col0 + t * 16 + lrow;
    const float bv = (bias && col < nbias) ? bias[col] : 0.0f;
#pragma unroll
    for (int v = 0; v < 8; ++v) {
      const int r = mb * 128 + wid * 16 + v + 8 * half;
      const float val = acc[t][v] + bv;
      if (OUT_BF16_RELU)
        ((__bf16*)Cv)[(size_t)r * N + col] = (__bf16)fmaxf(val, 0.0f);
      else
        ((float*)Cv)[(size_t)r * N + col] = val;
    }
  }
}

// ---------------------------------------------------------------------------
__global__ void k_convert(const float* __restrict__ src, __bf16* __restrict__ dst, int n) {
  int i = blockIdx.x * blockDim.x + threadIdx.x;
  if (i < n) dst[i] = (__bf16)src[i];
}

// dst[n*Kd + k] = (n < Nsrc) ? src[k*src_ld + n] : 0   for n < Npad, k < Kd
__global__ void k_transpose_conv(const float* __restrict__ src, __bf16* __restrict__ dst,
                                 int Kd, int Nsrc, int Npad, int src_ld) {
  int i = blockIdx.x * blockDim.x + threadIdx.x;
  if (i >= Npad * Kd) return;
  int n = i / Kd;
  int k = i - n * Kd;
  dst[(size_t)n * Kd + k] = (n < Nsrc) ? (__bf16)src[(size_t)k * src_ld + n] : (__bf16)0.0f;
}

__global__ void k_init(float* __restrict__ state, int* __restrict__ wprev,
                       float* __restrict__ out_states, float* __restrict__ out_logpt) {
  int i = blockIdx.x * blockDim.x + threadIdx.x;
  if (i >= 4096 * 64) return;
  int row = i >> 6, j = i & 63;
  state[i] = 0.0f;
  out_states[(size_t)row * (65 * 64) + j] = 0.0f;
  if (j == 0) { out_logpt[(size_t)row * 65 + 64] = 0.0f; wprev[row] = 0; }
}

// ---------------------------------------------------------------------------
// Per-row step: masked log-softmax (bwd + fwd), first-max argmax, state update,
// rank-1 update of h1pre. One 64-thread block per row.
// t in [0,63] = normal step; t == 64 = final pass (logpb_last only).
// ---------------------------------------------------------------------------
__global__ __launch_bounds__(64) void step_kernel(
    const float* __restrict__ logits,   // [B,192] (cols 0..64 = f, 65..128 = bk)
    float* __restrict__ state,          // [B,64]
    int* __restrict__ wprev,            // [B]
    const float* __restrict__ W1,       // [576,1024] f32
    float* __restrict__ h1pre,          // [B,1024] f32
    float* __restrict__ out_states, float* __restrict__ out_logpf,
    float* __restrict__ out_logpb, float* __restrict__ out_logpt, int t)
{
  const int row = blockIdx.x;
  const int j = threadIdx.x;     // 0..63
  __shared__ float s_st[64];
  __shared__ float s_red[64];
  __shared__ int   s_idx[64];

  const float* lg = logits + (size_t)row * 192;
  const float st = state[(size_t)row * 64 + j];
  s_st[j] = st;

  // ---- backward log-softmax over masked bk_logits ----
  const float bkm = (st != 0.0f) ? lg[65 + j] : NEG;
  s_red[j] = bkm; __syncthreads();
  for (int s = 32; s > 0; s >>= 1) { if (j < s) s_red[j] = fmaxf(s_red[j], s_red[j + s]); __syncthreads(); }
  const float bmax = s_red[0]; __syncthreads();
  s_red[j] = expf(bkm - bmax); __syncthreads();
  for (int s = 32; s > 0; s >>= 1) { if (j < s) s_red[j] += s_red[j + s]; __syncthreads(); }
  const float bsum = s_red[0]; __syncthreads();

  const int wp = wprev[row];
  const float bk_wp = (s_st[wp] != 0.0f) ? lg[65 + wp] : NEG;
  const float logpb_prev = bk_wp - bmax - logf(bsum);
  if (t >= 1 && j == 0) out_logpb[(size_t)row * 64 + (t - 1)] = logpb_prev;
  if (t >= 64) return;   // final pass: only logpb_last

  // ---- forward: mask = (state>0) for cols 0..63, col 64 always masked ----
  const float mf = (st > 0.0f) ? NEG : lg[j];
  s_red[j] = mf; __syncthreads();
  for (int s = 32; s > 0; s >>= 1) { if (j < s) s_red[j] = fmaxf(s_red[j], s_red[j + s]); __syncthreads(); }
  const float fmax = fmaxf(s_red[0], NEG); __syncthreads();
  s_red[j] = expf(mf - fmax); __syncthreads();
  for (int s = 32; s > 0; s >>= 1) { if (j < s) s_red[j] += s_red[j + s]; __syncthreads(); }
  const float fsum = s_red[0] + expf(NEG - fmax); __syncthreads();
  const float lse = fmax + logf(fsum);

  // ---- first-max argmax over 64 masked forward logits ----
  s_red[j] = mf; s_idx[j] = j; __syncthreads();
  for (int s = 32; s > 0; s >>= 1) {
    if (j < s) {
      float a = s_red[j], b = s_red[j + s];
      int ia = s_idx[j], ib = s_idx[j + s];
      if (b > a || (b == a && ib < ia)) { s_red[j] = b; s_idx[j] = ib; }
    }
    __syncthreads();
  }
  const int w = s_idx[0];
  const float mfw = s_red[0]; __syncthreads();

  // ---- outputs + state update ----
  const float ns = st + ((j == w) ? 1.0f : 0.0f);
  state[(size_t)row * 64 + j] = ns;
  out_states[(size_t)row * (65 * 64) + (size_t)(t + 1) * 64 + j] = ns;
  if (j == 0) {
    out_logpf[(size_t)row * 64 + t] = mfw - lse;
    out_logpt[(size_t)row * 65 + t] = NEG - lse;
    wprev[row] = w;
  }
  // rank-1 layer-1 update: h1pre[row] += W1[512+w, :]
  const float* wrow = W1 + (size_t)(512 + w) * 1024;
  float* hp = h1pre + (size_t)row * 1024;
  for (int c = j; c < 1024; c += 64) hp[c] += wrow[c];
}

// ---------------------------------------------------------------------------
extern "C" void kernel_launch(void* const* d_in, const int* in_sizes, int n_in,
                              void* d_out, int out_size, void* d_ws, size_t ws_size,
                              hipStream_t stream)
{
  const float* x  = (const float*)d_in[0];   // [4096,512]
  const float* W1 = (const float*)d_in[1];   // [576,1024]
  const float* b1 = (const float*)d_in[2];   // [1024]
  const float* W2 = (const float*)d_in[3];   // [1024,1024]
  const float* b2 = (const float*)d_in[4];   // [1024]
  const float* W3 = (const float*)d_in[5];   // [1024,129]
  const float* b3 = (const float*)d_in[6];   // [129]

  const int B = 4096, DIN = 512, DH = 1024, NB = 64, DOUT = 129, DOUTP = 192;

  char* ws = (char*)d_ws;
  size_t off = 0;
  auto alloc = [&](size_t bytes) -> char* {
    char* p = ws + off; off += (bytes + 255) & ~(size_t)255; return p;
  };
  __bf16* xbf   = (__bf16*)alloc((size_t)B * DIN * 2);
  __bf16* W1T   = (__bf16*)alloc((size_t)DH * DIN * 2);
  __bf16* W2T   = (__bf16*)alloc((size_t)DH * DH * 2);
  __bf16* W3T   = (__bf16*)alloc((size_t)DOUTP * DH * 2);
  float*  h1pre = (float*)alloc((size_t)B * DH * 4);
  __bf16* h2    = (__bf16*)alloc((size_t)B * DH * 2);
  float*  lgts  = (float*)alloc((size_t)B * DOUTP * 4);
  float*  state = (float*)alloc((size_t)B * NB * 4);
  int*    wprev = (int*)alloc((size_t)B * 4);

  float* out_states = (float*)d_out;                    // [B,65,64]
  float* out_logpf  = out_states + (size_t)B * 65 * 64; // [B,64]
  float* out_logpb  = out_logpf + (size_t)B * 64;       // [B,64]
  float* out_logpt  = out_logpb + (size_t)B * 64;       // [B,65]

  // prep: converts, transposes, init
  k_convert<<<(B * DIN + 255) / 256, 256, 0, stream>>>(x, xbf, B * DIN);
  k_transpose_conv<<<(DH * DIN + 255) / 256, 256, 0, stream>>>(W1, W1T, DIN, DH, DH, DH);
  k_transpose_conv<<<(DH * DH + 255) / 256, 256, 0, stream>>>(W2, W2T, DH, DH, DH, DH);
  k_transpose_conv<<<(DOUTP * DH + 255) / 256, 256, 0, stream>>>(W3, W3T, DH, DOUT, DOUTP, DOUT);
  k_init<<<(B * NB + 255) / 256, 256, 0, stream>>>(state, wprev, out_states, out_logpt);

  // h1pre = x @ W1[:512] + b1   (bf16 A, f32 out)
  // blocks = (4096/128) mblocks * (1024/64) nstrips = 32*16 = 512
  gemm_wmma_lds<false, false><<<512, 256, 0, stream>>>(
      xbf, W1T, b1, DH, h1pre, B, DH, DIN);

  for (int t = 0; t <= 64; ++t) {
    // h2 = relu(relu(h1pre) @ W2 + b2)   (fused relu+cvt A, bf16 relu out)
    gemm_wmma_lds<true, true><<<512, 256, 0, stream>>>(
        h1pre, W2T, b2, DH, h2, B, DH, DH);
    // logits = h2 @ W3 + b3  (N padded to 192; bias guarded to 129)
    gemm_wmma_lds<false, false><<<32 * 3, 256, 0, stream>>>(
        h2, W3T, b3, DOUT, lgts, B, DOUTP, DH);
    // per-row masked softmax / argmax / state + h1pre update
    step_kernel<<<B, 64, 0, stream>>>(
        lgts, state, wprev, W1, h1pre,
        out_states, out_logpf, out_logpb, out_logpt, t);
  }
}
